// GraphAutoEncoder_51496657879185
// MI455X (gfx1250) — matrix operand order repaired and checked by reading, not compiled, not run on hardware
//
#include <hip/hip_runtime.h>
#include <math.h>

#define N_NODES 100000
#define OUT_CH  128
#define N_EDGES 625000
#define EPS_F   1e-10f

typedef __attribute__((ext_vector_type(16))) _Float16 v16h;
typedef __attribute__((ext_vector_type(8)))  float    v8f;

// ---------------------------------------------------------------------------
// Kernel 1: zero the scatter accumulator (ptr) in workspace (float4 stores)
// ---------------------------------------------------------------------------
__global__ void gae_zero_kernel(float4* __restrict__ p, int n4) {
    int i = blockIdx.x * blockDim.x + threadIdx.x;
    if (i < n4) p[i] = make_float4(0.f, 0.f, 0.f, 0.f);
}

// ---------------------------------------------------------------------------
// Kernel 2: one-time weight conversion f32 -> f16 (row-major [128][128] x2).
// 8192 threads, each converts 4 elements. Runs once per launch; trivial cost.
// ---------------------------------------------------------------------------
__global__ void gae_wprep_kernel(const float* __restrict__ loc_w,
                                 const float* __restrict__ std_w,
                                 _Float16* __restrict__ wl16,
                                 _Float16* __restrict__ ws16) {
    int i = blockIdx.x * blockDim.x + threadIdx.x;   // 0..8191
    const int n = (OUT_CH * OUT_CH) / 4;             // 4096 float4 per head
    const float4* src;
    _Float16* dst;
    int j;
    if (i < n) { src = (const float4*)loc_w; dst = wl16; j = i; }
    else       { src = (const float4*)std_w; dst = ws16; j = i - n; }
    float4 v = src[j];
    dst[4 * j + 0] = (_Float16)v.x;
    dst[4 * j + 1] = (_Float16)v.y;
    dst[4 * j + 2] = (_Float16)v.z;
    dst[4 * j + 3] = (_Float16)v.w;
}

// ---------------------------------------------------------------------------
// Kernel 3: message = emb[sidx] * enorm ; scatter_add into ptr[tidx]
// One wave32 per edge, one float4 (4 channels) per lane -> 128 channels.
// ---------------------------------------------------------------------------
__global__ void gae_scatter_kernel(const float* __restrict__ emb,
                                   const int*   __restrict__ sidx,
                                   const int*   __restrict__ tidx,
                                   const float* __restrict__ enorm,
                                   float*       __restrict__ ptrm) {
    int gid  = blockIdx.x * blockDim.x + threadIdx.x;
    int e    = gid >> 5;
    int lane = gid & 31;
    if (e >= N_EDGES) return;
    int   s = sidx[e];
    int   t = tidx[e];
    float w = enorm[e];
    float4 v = ((const float4*)(emb + (size_t)s * OUT_CH))[lane];
    float* dst = ptrm + (size_t)t * OUT_CH + lane * 4;
    atomicAdd(dst + 0, v.x * w);
    atomicAdd(dst + 1, v.y * w);
    atomicAdd(dst + 2, v.z * w);
    atomicAdd(dst + 3, v.w * w);
}

// ---------------------------------------------------------------------------
// WMMA fragment loaders (wave32, v_wmma_f32_16x16x32_f16; contiguous K order used for both A and B)
// A 16x32 f16: lane m+16g, slots 0..15 -> K=kb+16g+h
// B 32x16 f16: lane n+16g, slots 0..15 -> K=kb+16g+h  (col n of B = row n of W)
// ---------------------------------------------------------------------------
__device__ __forceinline__ v16h gae_load_fragA(const float* __restrict__ row,
                                               int kb, int g) {
    const float4* p0 = (const float4*)(row + kb + 16 * g);
    const float4* p1 = (const float4*)(row + kb + 16 * g + 8);
    float4 a0 = p0[0], a1 = p0[1];
    float4 b0 = p1[0], b1 = p1[1];
    v16h r;
    r[0]  = (_Float16)a0.x; r[1]  = (_Float16)a0.y;
    r[2]  = (_Float16)a0.z; r[3]  = (_Float16)a0.w;
    r[4]  = (_Float16)a1.x; r[5]  = (_Float16)a1.y;
    r[6]  = (_Float16)a1.z; r[7]  = (_Float16)a1.w;
    r[8]  = (_Float16)b0.x; r[9]  = (_Float16)b0.y;
    r[10] = (_Float16)b0.z; r[11] = (_Float16)b0.w;
    r[12] = (_Float16)b1.x; r[13] = (_Float16)b1.y;
    r[14] = (_Float16)b1.z; r[15] = (_Float16)b1.w;
    return r;
}

// B fragment straight from pre-converted f16 weights: one aligned 32-byte load
__device__ __forceinline__ v16h gae_load_fragB16(const _Float16* __restrict__ row,
                                                 int kb, int g) {
    return *(const v16h*)(row + kb + 16 * g);
}

// ---------------------------------------------------------------------------
// Kernel 4: loc = ptr @ loc_w.T + loc_b ; std = softplus(ptr @ std_w.T + std_b)+eps
// Block = 256 threads (8 waves). Block -> one 16-row M tile (100000 = 6250*16).
// Wave w -> N tile w (16 cols). Each wave computes both heads, reusing A frags.
// K = 128 accumulated as 4 x v_wmma_f32_16x16x32_f16. No divergence: EXEC all-1.
// ---------------------------------------------------------------------------
__global__ void gae_gemm_heads_kernel(const float* __restrict__ ptrm,
                                      const _Float16* __restrict__ wl16,
                                      const _Float16* __restrict__ ws16,
                                      const float* __restrict__ loc_b,
                                      const float* __restrict__ std_b,
                                      float* __restrict__ loc_out,
                                      float* __restrict__ std_out) {
    const int lane  = threadIdx.x & 31;
    const int wave  = threadIdx.x >> 5;   // 0..7 -> N tile
    const int mbase = blockIdx.x * 16;
    const int nbase = wave * 16;
    const int lm    = lane & 15;          // A row / B col / D col within tile
    const int g     = lane >> 4;          // lane group

    // A fragments for 16 rows x 128 K (shared by both heads)
    const float* arow = ptrm + (size_t)(mbase + lm) * OUT_CH;
    v16h a0 = gae_load_fragA(arow, 0,  g);
    v16h a1 = gae_load_fragA(arow, 32, g);
    v16h a2 = gae_load_fragA(arow, 64, g);
    v16h a3 = gae_load_fragA(arow, 96, g);

    // ---- loc head ----
    {
        const _Float16* wrow = wl16 + (size_t)(nbase + lm) * OUT_CH;
        v8f acc = {};
        acc = __builtin_amdgcn_wmma_f32_16x16x32_f16(false, a0, false,
                gae_load_fragB16(wrow, 0,  g), (short)0, acc, false, false);
        acc = __builtin_amdgcn_wmma_f32_16x16x32_f16(false, a1, false,
                gae_load_fragB16(wrow, 32, g), (short)0, acc, false, false);
        acc = __builtin_amdgcn_wmma_f32_16x16x32_f16(false, a2, false,
                gae_load_fragB16(wrow, 64, g), (short)0, acc, false, false);
        acc = __builtin_amdgcn_wmma_f32_16x16x32_f16(false, a3, false,
                gae_load_fragB16(wrow, 96, g), (short)0, acc, false, false);
        const float bias = loc_b[nbase + lm];
#pragma unroll
        for (int v = 0; v < 8; ++v) {
            int r = mbase + v + 8 * g;
            loc_out[(size_t)r * OUT_CH + nbase + lm] = acc[v] + bias;
        }
    }

    // ---- std head ----
    {
        const _Float16* wrow = ws16 + (size_t)(nbase + lm) * OUT_CH;
        v8f acc = {};
        acc = __builtin_amdgcn_wmma_f32_16x16x32_f16(false, a0, false,
                gae_load_fragB16(wrow, 0,  g), (short)0, acc, false, false);
        acc = __builtin_amdgcn_wmma_f32_16x16x32_f16(false, a1, false,
                gae_load_fragB16(wrow, 32, g), (short)0, acc, false, false);
        acc = __builtin_amdgcn_wmma_f32_16x16x32_f16(false, a2, false,
                gae_load_fragB16(wrow, 64, g), (short)0, acc, false, false);
        acc = __builtin_amdgcn_wmma_f32_16x16x32_f16(false, a3, false,
                gae_load_fragB16(wrow, 96, g), (short)0, acc, false, false);
        const float bias = std_b[nbase + lm];
#pragma unroll
        for (int v = 0; v < 8; ++v) {
            int r = mbase + v + 8 * g;
            float x  = acc[v] + bias;
            float sp = (x > 20.0f) ? x : log1pf(expf(x));   // softplus
            std_out[(size_t)r * OUT_CH + nbase + lm] = sp + EPS_F;
        }
    }
}

// ---------------------------------------------------------------------------
// Kernel 5: logits[e] = dot(loc[sidx[e]], loc[tidx[e]]) over 128 channels.
// One wave32 per edge, float4 per lane, shfl_xor wave reduction.
// loc (51.2 MB) is L2-resident on MI455X (192 MB L2).
// ---------------------------------------------------------------------------
__global__ void gae_logits_kernel(const float* __restrict__ z,
                                  const int*   __restrict__ sidx,
                                  const int*   __restrict__ tidx,
                                  float*       __restrict__ logits) {
    int gid  = blockIdx.x * blockDim.x + threadIdx.x;
    int e    = gid >> 5;
    int lane = gid & 31;
    if (e >= N_EDGES) return;
    int s = sidx[e];
    int t = tidx[e];
    float4 va = ((const float4*)(z + (size_t)s * OUT_CH))[lane];
    float4 vb = ((const float4*)(z + (size_t)t * OUT_CH))[lane];
    float d = va.x * vb.x + va.y * vb.y + va.z * vb.z + va.w * vb.w;
#pragma unroll
    for (int off = 16; off > 0; off >>= 1)
        d += __shfl_xor(d, off, 32);
    if (lane == 0) logits[e] = d;
}

// ---------------------------------------------------------------------------
// Launch
// ---------------------------------------------------------------------------
extern "C" void kernel_launch(void* const* d_in, const int* in_sizes, int n_in,
                              void* d_out, int out_size, void* d_ws, size_t ws_size,
                              hipStream_t stream) {
    const float* emb    = (const float*)d_in[0];
    const int*   eidx   = (const int*)  d_in[1];   // [2, E] int32
    const float* enorm  = (const float*)d_in[2];
    const float* loc_w  = (const float*)d_in[3];
    const float* loc_b  = (const float*)d_in[4];
    const float* std_w  = (const float*)d_in[5];
    const float* std_b  = (const float*)d_in[6];

    const int* sidx = eidx;            // row 0
    const int* tidx = eidx + N_EDGES;  // row 1

    float* loc_out = (float*)d_out;                                   // [N, C]
    float* std_out = loc_out + (size_t)N_NODES * OUT_CH;              // [N, C]
    float* logits  = std_out + (size_t)N_NODES * OUT_CH;              // [E]

    // Workspace layout: [ptr f32 N*C][loc_w f16 128*128][std_w f16 128*128]
    float*    ptrm = (float*)d_ws;
    _Float16* wl16 = (_Float16*)(ptrm + (size_t)N_NODES * OUT_CH);
    _Float16* ws16 = wl16 + OUT_CH * OUT_CH;

    // 1) zero accumulator
    {
        int n4 = (N_NODES * OUT_CH) / 4;
        gae_zero_kernel<<<(n4 + 255) / 256, 256, 0, stream>>>((float4*)ptrm, n4);
    }
    // 2) weights f32 -> f16 (once)
    {
        int threads = 2 * (OUT_CH * OUT_CH) / 4;   // 8192
        gae_wprep_kernel<<<threads / 256, 256, 0, stream>>>(loc_w, std_w, wl16, ws16);
    }
    // 3) gather-scale-scatter (wave per edge)
    {
        long long threads = (long long)N_EDGES * 32;
        int blocks = (int)((threads + 255) / 256);
        gae_scatter_kernel<<<blocks, 256, 0, stream>>>(emb, sidx, tidx, enorm, ptrm);
    }
    // 4) dual-head WMMA GEMM (+bias, softplus)
    {
        gae_gemm_heads_kernel<<<N_NODES / 16, 256, 0, stream>>>(
            ptrm, wl16, ws16, loc_b, std_b, loc_out, std_out);
    }
    // 5) per-edge dot-product logits (wave per edge)
    {
        long long threads = (long long)N_EDGES * 32;
        int blocks = (int)((threads + 255) / 256);
        gae_logits_kernel<<<blocks, 256, 0, stream>>>(loc_out, sidx, tidx, logits);
    }
}
